// MultiHeadAttention_42064909697227
// MI455X (gfx1250) — matrix operand order repaired and checked by reading, not compiled, not run on hardware
//
#include <hip/hip_runtime.h>

// Problem constants (match reference)
#define DIMD 1024
#define BATCH 2
#define SEQ 2048
#define NH 16
#define HDIM 64

typedef __attribute__((ext_vector_type(16))) __bf16 bf16x16;
typedef __attribute__((ext_vector_type(16))) unsigned short u16x16;
typedef __attribute__((ext_vector_type(8))) float f32x8;
typedef __attribute__((address_space(3))) unsigned short lds_u16_t;

struct uint4x2 { uint4 x, y; };
static_assert(sizeof(bf16x16) == 32, "bf16x16 must be 32B");
static_assert(sizeof(uint4x2) == 32, "uint4x2 must be 32B");

// f32 -> bf16 bits via native conversion (v_cvt_pk_bf16_f32 on gfx1250)
__device__ __forceinline__ unsigned short f2bf(float f) {
  return __builtin_bit_cast(unsigned short, (__bf16)f);
}

// two 16B loads -> one 16-element bf16 fragment (elements 0..7 from p0, 8..15 from p1)
__device__ __forceinline__ bf16x16 ld_frag2(const unsigned short* p0, const unsigned short* p1) {
  uint4x2 t;
  t.x = *reinterpret_cast<const uint4*>(p0);
  t.y = *reinterpret_cast<const uint4*>(p1);
  return __builtin_bit_cast(bf16x16, t);
}

__device__ __forceinline__ f32x8 wmma_bf16(bf16x16 a, bf16x16 b, f32x8 c) {
  return __builtin_amdgcn_wmma_f32_16x16x32_bf16(false, a, false, b, (short)0, c, false, false);
}

// XOR-lane shuffle via ds_swizzle (group-of-32: offset = {xor[14:10], or[9:5], and[4:0]})
template <int XORM>
__device__ __forceinline__ float swz_xor(float v) {
  int i = __builtin_bit_cast(int, v);
  i = __builtin_amdgcn_ds_swizzle(i, (XORM << 10) | 0x1f);
  return __builtin_bit_cast(float, i);
}

// async-copy one contiguous 4KB tile (32 lanes x 8 chunks x 16B) global -> LDS
__device__ __forceinline__ void async_copy_4k(unsigned ldsBase, const unsigned short* gptr, int lane) {
  const unsigned long long g =
      (unsigned long long)(const void*)gptr + (unsigned)lane * 16u;
  const unsigned l = ldsBase + (unsigned)lane * 16u;
  #pragma unroll
  for (int c = 0; c < 8; ++c) {
    asm volatile("global_load_async_to_lds_b128 %0, %1, off"
                 :: "v"(l + (unsigned)(c * 512)), "v"(g + (unsigned long long)(c * 512))
                 : "memory");
  }
}

// ---------------------------------------------------------------------------
// Kernel 1: convert + transpose the four DxD weight matrices to bf16.
// Output layout: wts[w][n][k] = W_w[k][n], w in {q,k,v,o}
// ---------------------------------------------------------------------------
__global__ void prep_weights_k(const float* __restrict__ Wq, const float* __restrict__ Wk,
                               const float* __restrict__ Wv, const float* __restrict__ Wo,
                               unsigned short* __restrict__ out) {
  int idx = blockIdx.x * 256 + threadIdx.x;      // 0 .. 4*2^20-1
  int w   = idx >> 20;                           // D*D == 1<<20
  int rem = idx & ((1 << 20) - 1);
  int n = rem >> 10;
  int k = rem & 1023;
  const float* src = (w == 0) ? Wq : (w == 1) ? Wk : (w == 2) ? Wv : Wo;
  out[idx] = f2bf(src[k * DIMD + n]);
}

// ---------------------------------------------------------------------------
// Kernel 2: QKV projections. grid = (128, 4, 3), block = 256 (8 waves).
// Each wave: 16x64 output tile, K-loop step 32, v_wmma_f32_16x16x32_bf16.
// Q,K -> [B,H,S,HD] bf16;  V -> [B,H,HD,S] bf16 (transposed for P@V).
// ---------------------------------------------------------------------------
__global__ void qkv_proj_k(const float* __restrict__ Xq, const float* __restrict__ Xk,
                           const float* __restrict__ Xv, const unsigned short* __restrict__ wts,
                           const float* __restrict__ bq, const float* __restrict__ bk,
                           const float* __restrict__ bv, unsigned short* __restrict__ Qo,
                           unsigned short* __restrict__ Ko, unsigned short* __restrict__ Vo) {
  const int z = blockIdx.z;
  const float* X = (z == 0) ? Xq : (z == 1) ? Xk : Xv;
  const float* bias = (z == 0) ? bq : (z == 1) ? bk : bv;
  const unsigned short* Wt = wts + (size_t)z * (DIMD * DIMD);

  const int wave = threadIdx.x >> 5;
  const int lane = threadIdx.x & 31;
  const int hl = lane >> 4;       // half-wave selector
  const int ln = lane & 15;
  const int mi = wave & 1, ni = wave >> 1;
  const int row0 = blockIdx.x * 32 + mi * 16;
  const int col0 = blockIdx.y * 256 + ni * 64;

  f32x8 c[4] = {};
  const float* xr = X + (size_t)(row0 + ln) * DIMD;

  for (int k = 0; k < DIMD; k += 32) {
    // A fragment: lane row = ln; k-elems {o..o+7, o+16..o+23}, o = k + hl*8
    const int o0 = k + hl * 8;
    float4 f0 = *reinterpret_cast<const float4*>(xr + o0);
    float4 f1 = *reinterpret_cast<const float4*>(xr + o0 + 4);
    float4 f2 = *reinterpret_cast<const float4*>(xr + o0 + 16);
    float4 f3 = *reinterpret_cast<const float4*>(xr + o0 + 20);
    u16x16 au;
    au[0] = f2bf(f0.x);  au[1] = f2bf(f0.y);  au[2] = f2bf(f0.z);  au[3] = f2bf(f0.w);
    au[4] = f2bf(f1.x);  au[5] = f2bf(f1.y);  au[6] = f2bf(f1.z);  au[7] = f2bf(f1.w);
    au[8] = f2bf(f2.x);  au[9] = f2bf(f2.y);  au[10] = f2bf(f2.z); au[11] = f2bf(f2.w);
    au[12] = f2bf(f3.x); au[13] = f2bf(f3.y); au[14] = f2bf(f3.z); au[15] = f2bf(f3.w);
    bf16x16 a = __builtin_bit_cast(bf16x16, au);

    #pragma unroll
    for (int t = 0; t < 4; ++t) {
      // B fragment: lane col n; k-elems {o..o+7, o+16..o+23}, o = k + hl*8
      const unsigned short* wr = Wt + (size_t)(col0 + t * 16 + ln) * DIMD + k + hl * 8;
      c[t] = wmma_bf16(a, ld_frag2(wr, wr + 16), c[t]);
    }
  }

  // epilogue: C element (m = r + 8*hl, n = col + ln); z-switch hoisted out
  const int b = (row0 >> 11);            // tile never straddles batch boundary
  const int s0r = row0 & (SEQ - 1);
  if (z == 2) {
    #pragma unroll
    for (int t = 0; t < 4; ++t) {
      int n = col0 + t * 16 + ln;
      int h = n >> 6, hd = n & (HDIM - 1);
      float bs = bias[n];
      #pragma unroll
      for (int r = 0; r < 8; ++r) {
        int s = s0r + r + hl * 8;
        Vo[(size_t)((b * NH + h) * HDIM + hd) * SEQ + s] = f2bf(c[t][r] + bs);
      }
    }
  } else {
    unsigned short* O = (z == 0) ? Qo : Ko;
    #pragma unroll
    for (int t = 0; t < 4; ++t) {
      int n = col0 + t * 16 + ln;
      int h = n >> 6, hd = n & (HDIM - 1);
      float bs = bias[n];
      #pragma unroll
      for (int r = 0; r < 8; ++r) {
        int s = s0r + r + hl * 8;
        O[(size_t)((b * NH + h) * SEQ + s) * HDIM + hd] = f2bf(c[t][r] + bs);
      }
    }
  }
}

// ---------------------------------------------------------------------------
// Kernel 3: causal flash attention. grid = (16, 32), block = 256 (8 waves).
// Wave handles 16 query rows, streams 32-key tiles with online softmax.
// K tiles double-buffered in LDS via global_load_async_to_lds_b128:
// tile j+1's async group is issued before computing tile j, then
// s_wait_asynccnt 8 (in-order completion => tile j fully landed).
// ---------------------------------------------------------------------------
__global__ void attn_k(const unsigned short* __restrict__ Qg, const unsigned short* __restrict__ Kg,
                       const unsigned short* __restrict__ Vg, unsigned short* __restrict__ Cg) {
  __shared__ __align__(16) unsigned short ldsP[8][16 * 32];       // per-wave P tile (1 KB)
  __shared__ __align__(16) unsigned short ldsK[8][2][32 * 64];    // per-wave 2x K tile (8 KB)

  const int wave = threadIdx.x >> 5;
  const int lane = threadIdx.x & 31;
  const int hl = lane >> 4;
  const int ln = lane & 15;
  const int bh = blockIdx.y;
  const int b = bh >> 4, h = bh & 15;
  const int q0 = (blockIdx.x * 8 + wave) * 16;

  const unsigned short* Q = Qg + (size_t)bh * SEQ * HDIM;
  const unsigned short* K = Kg + (size_t)bh * SEQ * HDIM;
  const unsigned short* Vt = Vg + (size_t)bh * HDIM * SEQ;   // [HD][S]

  // Q A-fragments (row = ln): d in {0..31} and {32..63}
  const unsigned short* qrow = Q + (size_t)(q0 + ln) * HDIM;
  const bf16x16 qa0 = ld_frag2(qrow + hl * 16, qrow + hl * 16 + 8);
  const bf16x16 qa1 = ld_frag2(qrow + 32 + hl * 16, qrow + 32 + hl * 16 + 8);

  f32x8 o0 = {}, o1 = {}, o2 = {}, o3 = {};
  float mrow[8], lrow[8];
  #pragma unroll
  for (int r = 0; r < 8; ++r) { mrow[r] = -1e30f; lrow[r] = 0.0f; }

  unsigned short* myl = &ldsP[wave][0];
  unsigned short* myk = &ldsK[wave][0][0];
  // 32-bit LDS byte offset of this wave's K double-buffer (addrspace(3) view)
  const unsigned kOff = (unsigned)(unsigned long long)(lds_u16_t*)myk;

  const int ntiles = (q0 + 47) >> 5;   // ceil((q0+16)/32): causal bound

  // prologue: stage tile 0 into buffer 0
  async_copy_4k(kOff, K, lane);

  for (int j = 0; j < ntiles; ++j) {
    const int ks = j * 32;
    const int buf = j & 1;
    f32x8 zero = {};

    // prior iteration's LDS reads (of the buffer we are about to refill) done
    asm volatile("s_wait_dscnt 0" ::: "memory");

    if (j + 1 < ntiles) {
      // issue next tile into the other buffer, prefetch the one after to L2
      async_copy_4k(kOff + (unsigned)((buf ^ 1) * 4096), K + (size_t)(ks + 32) * HDIM, lane);
      __builtin_prefetch(K + (size_t)(ks + 64 + lane) * HDIM, 0, 3);
      __builtin_prefetch(Vt + (size_t)(2 * lane) * SEQ + ks + 32, 0, 3);
      // 8 ops of tile j+1 may remain in flight; in-order => tile j complete
      asm volatile("s_wait_asynccnt 0x8" ::: "memory");
    } else {
      asm volatile("s_wait_asynccnt 0x0" ::: "memory");
    }

    // --- scores = Q @ K^T for 32 keys (two 16-col tiles), K frags from LDS ---
    const unsigned short* kb = myk + buf * (32 * 64);
    const unsigned short* kr0 = kb + (size_t)ln * HDIM;          // key ks+ln
    const unsigned short* kr1 = kb + (size_t)(16 + ln) * HDIM;   // key ks+16+ln
    f32x8 s0 = wmma_bf16(qa0, ld_frag2(kr0 + hl * 16, kr0 + hl * 16 + 8), zero);
    s0 = wmma_bf16(qa1, ld_frag2(kr0 + 32 + hl * 16, kr0 + 32 + hl * 16 + 8), s0);
    f32x8 s1 = wmma_bf16(qa0, ld_frag2(kr1 + hl * 16, kr1 + hl * 16 + 8), zero);
    s1 = wmma_bf16(qa1, ld_frag2(kr1 + 32 + hl * 16, kr1 + 32 + hl * 16 + 8), s1);

    // --- scale, causal mask, online softmax (row = r + 8*hl, col = lane's n) ---
    float alpha[8];
    #pragma unroll
    for (int r = 0; r < 8; ++r) {
      int qi = q0 + r + hl * 8;
      float v0 = s0[r] * 0.125f;                 // 1/sqrt(64)
      float v1 = s1[r] * 0.125f;
      v0 = ((ks + ln) > qi) ? -1e30f : v0;
      v1 = ((ks + 16 + ln) > qi) ? -1e30f : v1;
      float mx = fmaxf(v0, v1);
      mx = fmaxf(mx, swz_xor<1>(mx));
      mx = fmaxf(mx, swz_xor<2>(mx));
      mx = fmaxf(mx, swz_xor<4>(mx));
      mx = fmaxf(mx, swz_xor<8>(mx));
      float mnew = fmaxf(mrow[r], mx);
      float al = __expf(mrow[r] - mnew);
      mrow[r] = mnew;
      alpha[r] = al;
      float p0 = __expf(v0 - mnew);
      float p1 = __expf(v1 - mnew);
      float rs = p0 + p1;
      rs += swz_xor<1>(rs);
      rs += swz_xor<2>(rs);
      rs += swz_xor<4>(rs);
      rs += swz_xor<8>(rs);
      lrow[r] = lrow[r] * al + rs;
      // stage P (bf16) to LDS in row-major [16][32] for layout conversion
      int m = r + hl * 8;
      myl[m * 32 + ln] = f2bf(p0);
      myl[m * 32 + 16 + ln] = f2bf(p1);
    }
    #pragma unroll
    for (int r = 0; r < 8; ++r) {
      o0[r] *= alpha[r]; o1[r] *= alpha[r]; o2[r] *= alpha[r]; o3[r] *= alpha[r];
    }

    asm volatile("s_wait_dscnt 0" ::: "memory");

    // reload P as an A-fragment (row = ln; 16 contiguous k-elems starting at hl*16)
    const unsigned short* pr = myl + ln * 32 + hl * 16;
    bf16x16 pa = ld_frag2(pr, pr + 8);

    // --- O += P @ V : four 16-wide d tiles, V transposed so keys contiguous ---
    {
      const unsigned short* p = Vt + (size_t)(ln) * SEQ + ks + hl * 16;
      o0 = wmma_bf16(pa, ld_frag2(p, p + 8), o0);
    }
    {
      const unsigned short* p = Vt + (size_t)(16 + ln) * SEQ + ks + hl * 16;
      o1 = wmma_bf16(pa, ld_frag2(p, p + 8), o1);
    }
    {
      const unsigned short* p = Vt + (size_t)(32 + ln) * SEQ + ks + hl * 16;
      o2 = wmma_bf16(pa, ld_frag2(p, p + 8), o2);
    }
    {
      const unsigned short* p = Vt + (size_t)(48 + ln) * SEQ + ks + hl * 16;
      o3 = wmma_bf16(pa, ld_frag2(p, p + 8), o3);
    }
  }

  // epilogue: normalize and write context (concat layout [B,S,D] bf16)
  #pragma unroll
  for (int r = 0; r < 8; ++r) {
    int srow = q0 + r + hl * 8;
    float inv = 1.0f / lrow[r];
    size_t base = (size_t)(b * SEQ + srow) * DIMD + h * HDIM + ln;
    Cg[base + 0]  = f2bf(o0[r] * inv);
    Cg[base + 16] = f2bf(o1[r] * inv);
    Cg[base + 32] = f2bf(o2[r] * inv);
    Cg[base + 48] = f2bf(o3[r] * inv);
  }
}

// ---------------------------------------------------------------------------
// Kernel 4: output projection. grid = (128, 4), block = 256 (8 waves).
// Ctx(bf16) @ Wo_t(bf16) + bo -> f32 out.
// ---------------------------------------------------------------------------
__global__ void out_proj_k(const unsigned short* __restrict__ Ctx, const unsigned short* __restrict__ Wt,
                           const float* __restrict__ bo, float* __restrict__ out) {
  const int wave = threadIdx.x >> 5;
  const int lane = threadIdx.x & 31;
  const int hl = lane >> 4;
  const int ln = lane & 15;
  const int mi = wave & 1, ni = wave >> 1;
  const int row0 = blockIdx.x * 32 + mi * 16;
  const int col0 = blockIdx.y * 256 + ni * 64;

  f32x8 c[4] = {};
  const unsigned short* ar = Ctx + (size_t)(row0 + ln) * DIMD;

  for (int k = 0; k < DIMD; k += 32) {
    const unsigned short* ap = ar + k + hl * 16;
    bf16x16 a = ld_frag2(ap, ap + 8);
    #pragma unroll
    for (int t = 0; t < 4; ++t) {
      const unsigned short* wr = Wt + (size_t)(col0 + t * 16 + ln) * DIMD + k + hl * 16;
      c[t] = wmma_bf16(a, ld_frag2(wr, wr + 8), c[t]);
    }
  }

  #pragma unroll
  for (int t = 0; t < 4; ++t) {
    int n = col0 + t * 16 + ln;
    float bs = bo[n];
    #pragma unroll
    for (int r = 0; r < 8; ++r) {
      int row = row0 + r + hl * 8;
      out[(size_t)row * DIMD + n] = c[t][r] + bs;
    }
  }
}

// ---------------------------------------------------------------------------
// Host launch
// ---------------------------------------------------------------------------
extern "C" void kernel_launch(void* const* d_in, const int* in_sizes, int n_in,
                              void* d_out, int out_size, void* d_ws, size_t ws_size,
                              hipStream_t stream) {
  (void)in_sizes; (void)n_in; (void)out_size; (void)ws_size;

  const float* q  = (const float*)d_in[0];
  const float* k  = (const float*)d_in[1];
  const float* v  = (const float*)d_in[2];
  // d_in[3] = causal mask (bool), structure known at compile time -> unused
  const float* Wq = (const float*)d_in[4];
  const float* bq = (const float*)d_in[5];
  const float* Wk = (const float*)d_in[6];
  const float* bk = (const float*)d_in[7];
  const float* Wv = (const float*)d_in[8];
  const float* bv = (const float*)d_in[9];
  const float* Wo = (const float*)d_in[10];
  const float* bo = (const float*)d_in[11];
  float* out = (float*)d_out;

  const size_t MAT = (size_t)DIMD * DIMD;          // 1M elems
  const size_t ACT = (size_t)BATCH * SEQ * DIMD;   // 4M elems
  unsigned short* wts = (unsigned short*)d_ws;     // 4*MAT bf16 (q,k,v,o transposed)
  unsigned short* Qb  = wts + 4 * MAT;             // [B,H,S,HD]
  unsigned short* Kb  = Qb + ACT;                  // [B,H,S,HD]
  unsigned short* Vtb = Kb + ACT;                  // [B,H,HD,S]
  unsigned short* Ctx = Vtb + ACT;                 // [B,S,D]

  prep_weights_k<<<dim3((unsigned)(4 * MAT / 256)), 256, 0, stream>>>(Wq, Wk, Wv, Wo, wts);
  qkv_proj_k<<<dim3(128, 4, 3), 256, 0, stream>>>(q, k, v, wts, bq, bk, bv, Qb, Kb, Vtb);
  attn_k<<<dim3(16, 32), 256, 0, stream>>>(Qb, Kb, Vtb, Ctx);
  out_proj_k<<<dim3(128, 4), 256, 0, stream>>>(Ctx, wts + 3 * MAT, bo, out);
}
